// QRUBlock_79731772883414
// MI455X (gfx1250) — matrix operand order repaired and checked by reading, not compiled, not run on hardware
//
#include <hip/hip_runtime.h>
#include <math.h>

// ---------------------------------------------------------------------------
// QRU block for MI455X (gfx1250, wave32, WMMA).
//   conv1(1->32ch, 3^3) as WMMA implicit GEMM (M=32,K=27->32,N=x)
//   -> BN stats -> BN+tanh/sigmoid+temporal scan
//   -> fused conv2/conv3 (M=2 rows in a 16-row tile, K'=27*16=432->448)
//   -> BN stats -> sigmoid(+2)+eps / identity -> d_out.
// Bandwidth-bound (~4 GFLOP vs a few hundred MB @ 23.3 TB/s): the two big
// intermediates are f16 *channel-last* (NDHWC). h_time lives in a
// zero-padded buffer (T+2)x(H+2)x(W+2) so conv2's B fragments -- with K'
// ordered tap-major, K' = tap*16 + ci -- are unconditional contiguous 32B
// reads of h16p: no im2col staging, no halo selects in the WMMA loop.
// conv1 stages its halo patch with global_load_async_to_lds (ASYNCcnt).
// ---------------------------------------------------------------------------

typedef __attribute__((ext_vector_type(16))) _Float16 v16h;
typedef __attribute__((ext_vector_type(8)))  _Float16 v8h;
typedef __attribute__((ext_vector_type(8)))  float    v8f;
typedef __attribute__((ext_vector_type(2)))  float    v2f;

#define BB   2
#define TT   31
#define HH   160
#define WW   160
#define HW   (HH * WW)          // 25600
#define NPOS (BB * TT * HW)     // 1,587,200 positions
#define HC   16
// padded h_time dims (one halo cell each side of t, y, x)
#define TP   (TT + 2)           // 33
#define HP   (HH + 2)           // 162
#define WP   (WW + 2)           // 162
#define H16P_ELEMS ((size_t)BB * TP * HP * WP * HC)   // 27,713,664 halfs

// ---------------- stats layout in workspace (floats) -----------------------
// [0..31]   gate_sum      [32..63]  gate_sq
// [64..95]  gate_mean     [96..127] gate_istd
// [128]=s_sum [129]=h_sum [130]=s_sq [131]=h_sq
// [132]=s_mean [133]=h_mean [134]=s_istd [135]=h_istd
#define STATS_FLOATS 256

__global__ void zero_stats_kernel(float* __restrict__ s) {
    s[threadIdx.x] = 0.0f;
}

// zero the padded h_time buffer (16B vector stores; scan rewrites interior)
__global__ __launch_bounds__(256) void zero_h16p_kernel(v8h* __restrict__ p, long n8) {
    const long i = (long)blockIdx.x * 256 + threadIdx.x;
    if (i < n8) p[i] = (v8h){};
}

// ---------------------------------------------------------------------------
// Pre-swizzle weights into per-lane WMMA A-fragment order (done once).
// A-frag (16-bit 16x32, contiguous k): lane(lo,hi), elem e -> K = kk*32 + 16*hi +
// e, row M = lo.
// gW1: conv1   [mt*16+lo][hi][16 e], K = tap (27 taps, padded to 32).
// gW2: conv2/3 [lo][hi][kk 0..13][16 e], K' = tap*16 + ci  (tap-major!).
//      row0 = w_scale over ci0-7, row1 = w_shift over ci8-15, rest 0.
// ---------------------------------------------------------------------------
__global__ void build_wfrag_kernel(const float* __restrict__ wG,
                                   const float* __restrict__ wS,
                                   const float* __restrict__ wH,
                                   _Float16* __restrict__ gW1,
                                   _Float16* __restrict__ gW2) {
    const int tid = threadIdx.x;
    for (int i = tid; i < 1024; i += 256) {
        const int e = i & 15, hi = (i >> 4) & 1, row = i >> 5;  // row = mt*16+lo
        const int k = 16 * hi + e;
        gW1[i] = (_Float16)(k < 27 ? wG[row * 27 + k] : 0.0f);
    }
    for (int i = tid; i < 7168; i += 256) {
        const int e = i & 15;
        int r = i >> 4;
        const int kk = r % 14; r /= 14;
        const int hi = r & 1;
        const int lo = r >> 1;
        const int k = kk * 32 + 16 * hi + e;  // K' index
        float v = 0.0f;
        if (k < 432) {
            const int tap = k >> 4, ci = k & 15;               // tap-major
            if (lo == 0 && ci < 8)       v = wS[ci * 27 + tap];
            else if (lo == 1 && ci >= 8) v = wH[(ci - 8) * 27 + tap];
        }
        gW2[i] = (_Float16)v;
    }
}

// ---------------------------------------------------------------------------
// conv1: one block per (b,t,y) row. 320 threads = 10 waves x 16 x-positions.
// Halo patch async-staged to LDS, im2col panel built in B-fragment order,
// 2 WMMAs per wave. Output: channel-last f16 gates, two 16B v8h stores/lane.
// ---------------------------------------------------------------------------
__global__ __launch_bounds__(320) void conv1_wmma_kernel(
    const float* __restrict__ in, const _Float16* __restrict__ gW1,
    _Float16* __restrict__ g16, float* __restrict__ g_sum, float* __restrict__ g_sq) {

    __shared__ __attribute__((aligned(32))) float    sPf[3 * 3 * 162]; // halo, x in [-1,160]
    __shared__ __attribute__((aligned(32))) _Float16 BmatT[160 * 32];  // [n][k] B panel
    __shared__ int   offTab[32];
    __shared__ float sSum[32], sSq[32];

    const int bid = blockIdx.x;
    const int y = bid % HH;
    const int t = (bid / HH) % TT;
    const int b = bid / (HH * TT);
    const int tid = threadIdx.x;

    if (tid < 32) {
        int v = -1;
        if (tid < 27) {
            const int dt = tid / 9, r = tid % 9;
            v = (dt * 3 + r / 3) * 162 + (r % 3);
        }
        offTab[tid] = v;
        sSum[tid] = 0.0f;
        sSq[tid] = 0.0f;
    }

    // stage halo patch via async global->LDS copies (ASYNCcnt); OOB -> 0
    for (int i = tid; i < 3 * 3 * 162; i += 320) {
        const int px = i % 162;
        const int dy = (i / 162) % 3;
        const int dt = i / (162 * 3);
        const int tt = t + dt - 1, yy = y + dy - 1, xx = px - 1;
        if ((unsigned)tt < TT && (unsigned)yy < HH && (unsigned)xx < WW) {
            const float* gp = &in[((size_t)(b * TT + tt) * HH + yy) * WW + xx];
            const unsigned lp = (unsigned)(size_t)&sPf[i];
            asm volatile("global_load_async_to_lds_b32 %0, %1, off"
                         :: "v"(lp), "v"(gp) : "memory");
        } else {
            sPf[i] = 0.0f;
        }
    }
    asm volatile("s_wait_asynccnt 0x0" ::: "memory");
    __syncthreads();

    // B panel in fragment order: BmatT[x][k], k contiguous per lane
    for (int i = tid; i < 160 * 32; i += 320) {
        const int x = i >> 5, k = i & 31;
        const int off = offTab[k];
        BmatT[i] = (off >= 0) ? (_Float16)sPf[off + x] : (_Float16)0.0f;
    }
    __syncthreads();

    const int wave = tid >> 5, lane = tid & 31;
    const int lo = lane & 15, hi = lane >> 4;
    const int xbase = wave * 16;

    const v16h a0 = *(const v16h*)&gW1[(lo * 2 + hi) * 16];
    const v16h a1 = *(const v16h*)&gW1[((16 + lo) * 2 + hi) * 16];
    const v16h bf = *(const v16h*)&BmatT[(xbase + lo) * 32 + 16 * hi];

    v8f c0 = {}, c1 = {};
    c0 = __builtin_amdgcn_wmma_f32_16x16x32_f16(false, a0, false, bf, (short)0, c0, false, false);
    c1 = __builtin_amdgcn_wmma_f32_16x16x32_f16(false, a1, false, bf, (short)0, c1, false, false);

    // channel-last epilogue: BN sums from f32 accumulators, then two 16B stores
    const int x = xbase + lo;
    const size_t pos = (((size_t)b * TT + t) * HH + y) * WW + x;
    v8h o0, o1;
#pragma unroll
    for (int j = 0; j < 8; ++j) {
        const int ch = j + 8 * hi;           // C/D layout: VGPR j -> M = j + 8*hi
        const float v0 = c0[j], v1 = c1[j];
        o0[j] = (_Float16)v0;
        o1[j] = (_Float16)v1;
        atomicAdd(&sSum[ch], v0);       atomicAdd(&sSq[ch], v0 * v0);
        atomicAdd(&sSum[ch + 16], v1);  atomicAdd(&sSq[ch + 16], v1 * v1);
    }
    _Float16* gp = &g16[pos * 32 + 8 * hi];
    *(v8h*)gp        = o0;   // channels 8*hi .. 8*hi+7
    *(v8h*)(gp + 16) = o1;   // channels 16+8*hi .. 16+8*hi+7
    __syncthreads();
    if (tid < 32) { atomicAdd(&g_sum[tid], sSum[tid]); atomicAdd(&g_sq[tid], sSq[tid]); }
}

__global__ void finalize_stats_kernel(const float* __restrict__ sum,
                                      const float* __restrict__ sq,
                                      float* __restrict__ mean,
                                      float* __restrict__ istd,
                                      int n, float inv_count) {
    const int i = threadIdx.x;
    if (i < n) {
        const float m = sum[i] * inv_count;
        const float v = sq[i] * inv_count - m * m;
        mean[i] = m;
        istd[i] = rsqrtf(v + 1e-5f);
    }
}

// ---------------------------------------------------------------------------
// BN + tanh/sigmoid + temporal scan over channel-last f16 gates.
// One thread per (b, y, x, c-half): two contiguous v8h loads per t (lane
// pairs cover full 64B lines), 8 recurrences in registers, one v8h store
// into the *padded* h16p interior.
// ---------------------------------------------------------------------------
__global__ __launch_bounds__(256) void scan_kernel(
    const _Float16* __restrict__ g16,
    const float* __restrict__ mean, const float* __restrict__ istd,
    const float* __restrict__ gamma, const float* __restrict__ beta,
    const int* __restrict__ rev, _Float16* __restrict__ h16p) {

    const int tid = threadIdx.x + blockIdx.x * 256;
    if (tid >= BB * 2 * HW) return;
    const int c8 = tid & 1;          // which 8-channel half
    const int p  = tid >> 1;         // b*HW + y*WW + x
    const int b  = p / HW;
    const int pr = p % HW;
    const int yq = pr / WW, xq = pr % WW;
    const int cb = c8 * 8;

    float mz[8], iz[8], gz[8], bz[8], mf[8], fi[8], gf[8], bfv[8];
#pragma unroll
    for (int j = 0; j < 8; ++j) {
        mz[j] = mean[cb + j];       iz[j] = istd[cb + j];
        gz[j] = gamma[cb + j];      bz[j] = beta[cb + j];
        mf[j] = mean[cb + j + HC];  fi[j] = istd[cb + j + HC];
        gf[j] = gamma[cb + j + HC]; bfv[j] = beta[cb + j + HC];
    }
    const int r = rev[0];

    float hv[8] = {};
    for (int s = 0; s < TT; ++s) {
        const int t = r ? (TT - 1 - s) : s;
        const size_t pos = ((size_t)b * TT + t) * HW + pr;
        const _Float16* gp = &g16[pos * 32 + cb];
        const v8h zv = *(const v8h*)gp;
        const v8h fv = *(const v8h*)(gp + HC);
        if (s + 1 < TT) {
            const int tn = r ? (TT - 2 - s) : (s + 1);
            __builtin_prefetch(&g16[(((size_t)b * TT + tn) * HW + pr) * 32], 0, 0);
        }
        v8h ho;
#pragma unroll
        for (int j = 0; j < 8; ++j) {
            const float z = tanhf(((float)zv[j] - mz[j]) * iz[j] * gz[j] + bz[j]);
            const float f = 1.0f / (1.0f + expf(-(((float)fv[j] - mf[j]) * fi[j] * gf[j] + bfv[j])));
            hv[j] = f * hv[j] + (1.0f - f) * z;
            ho[j] = (_Float16)hv[j];
        }
        const size_t posp = (((size_t)b * TP + (t + 1)) * HP + (yq + 1)) * WP + (xq + 1);
        *(v8h*)&h16p[posp * HC + cb] = ho;
    }
}

// ---------------------------------------------------------------------------
// Fused conv2/conv3 over padded channel-last f16 h, K' tap-major: a lane's
// B fragment (elem i <-> K = kk*32 + 16*hi + i, contiguous order as in A) is the 16 channels of
// h16p at tap (2kk+hi) and one x -- one unconditional contiguous 32B global
// load (padding guarantees in-bounds zeros; the K-pad tap reads element 0,
// which is pad). One block per (b,t,y) row: 320 threads = 10 waves x 16 x.
// Raw (scale,shift) interleaved: one 8B v2f store per lane.
// ---------------------------------------------------------------------------
__global__ __launch_bounds__(320) void conv2_wmma_kernel(
    const _Float16* __restrict__ h16p, const _Float16* __restrict__ gW2,
    float* __restrict__ raw,
    float* __restrict__ st /* = stats+128: [s_sum,h_sum,s_sq,h_sq] */) {

    __shared__ float sAcc[4];

    const int bid = blockIdx.x;
    const int y = bid % HH;
    const int t = (bid / HH) % TT;
    const int b = bid / (HH * TT);
    const int tid = threadIdx.x;
    if (tid < 4) sAcc[tid] = 0.0f;
    __syncthreads();

    const int wave = tid >> 5, lane = tid & 31;
    const int lo = lane & 15, hi = lane >> 4;
    const int xg = wave * 16 + lo;   // output x, 0..159

    // per-lane row bases into padded h16p, xg folded in; (t+dt-1+1) = t+dt
    int rp[9];
#pragma unroll
    for (int rid = 0; rid < 9; ++rid) {
        const int dt = rid / 3, dy = rid % 3;
        rp[rid] = (((b * TP + (t + dt)) * HP + (y + dy)) * WP + xg) * HC;
    }

    const _Float16* ap = &gW2[((lo * 2 + hi) * 14) * 16];
    v8f c = {};
#pragma unroll
    for (int kk = 0; kk < 14; ++kk) {
        const int tap0 = 2 * kk;          // hi==0 tap (always < 27)
        const int tap1 = 2 * kk + 1;      // hi==1 tap (27 => K-pad -> elem 0 = pad = 0)
        const int off0 = rp[tap0 / 3] + (tap0 % 3) * HC;
        const int off1 = (tap1 < 27) ? (rp[tap1 / 3] + (tap1 % 3) * HC) : 0;
        const int boff = hi ? off1 : off0;
        const v16h bfr = *(const v16h*)&h16p[boff];
        const v16h a   = *(const v16h*)(ap + kk * 16);
        c = __builtin_amdgcn_wmma_f32_16x16x32_f16(false, a, false, bfr, (short)0, c, false, false);
    }

    // M=0 (scale) / M=1 (shift) live in VGPR0/VGPR1 on lanes 0-15
    if (hi == 0) {
        const float vs = c[0], vh = c[1];
        const size_t idx = (((size_t)b * TT + t) * HH + y) * WW + xg;
        const v2f o = {vs, vh};
        *(v2f*)&raw[idx * 2] = o;
        atomicAdd(&sAcc[0], vs); atomicAdd(&sAcc[2], vs * vs);
        atomicAdd(&sAcc[1], vh); atomicAdd(&sAcc[3], vh * vh);
    }
    __syncthreads();
    if (tid < 4) atomicAdd(&st[tid], sAcc[tid]);
}

__global__ __launch_bounds__(256) void final_kernel(
    const float* __restrict__ raw,
    const float* __restrict__ st,   // stats+128
    const float* __restrict__ gS, const float* __restrict__ bS,
    const float* __restrict__ gH, const float* __restrict__ bH,
    float* __restrict__ out) {
    const int i = threadIdx.x + blockIdx.x * 256;
    if (i >= NPOS) return;
    const v2f rv = *(const v2f*)&raw[i * 2];
    const float vs = (rv.x - st[4]) * st[6] * gS[0] + bS[0];
    const float sc = 1.0f / (1.0f + expf(-(vs + 2.0f))) + 1e-4f;
    const float vh = (rv.y - st[5]) * st[7] * gH[0] + bH[0];
    out[i] = sc;
    out[NPOS + i] = vh;
}

extern "C" void kernel_launch(void* const* d_in, const int* in_sizes, int n_in,
                              void* d_out, int out_size, void* d_ws, size_t ws_size,
                              hipStream_t stream) {
    const float* inputs      = (const float*)d_in[0];
    const float* w_gate      = (const float*)d_in[1];
    const float* gamma_gate  = (const float*)d_in[2];
    const float* beta_gate   = (const float*)d_in[3];
    const float* w_scale     = (const float*)d_in[4];
    const float* gamma_scale = (const float*)d_in[5];
    const float* beta_scale  = (const float*)d_in[6];
    const float* w_shift     = (const float*)d_in[7];
    const float* gamma_shift = (const float*)d_in[8];
    const float* beta_shift  = (const float*)d_in[9];
    const int*   reverse     = (const int*)d_in[10];
    float* out = (float*)d_out;

    // workspace carve-up (all region sizes are multiples of 32B)
    float*    stats = (float*)d_ws;                                   // 256 f32
    _Float16* g16   = (_Float16*)(stats + STATS_FLOATS);              // NPOS*32 f16 (channel-last)
    _Float16* h16p  = g16 + (size_t)NPOS * 32;                        // padded channel-last h
    float*    raw   = (float*)(h16p + H16P_ELEMS);                    // NPOS*2 f32 (interleaved)
    _Float16* gW1   = (_Float16*)(raw + (size_t)NPOS * 2);            // 1024 halfs
    _Float16* gW2   = gW1 + 1024;                                     // 7168 halfs

    const float inv_count = 1.0f / (float)NPOS;
    const long  n8 = (long)(H16P_ELEMS / 8);

    zero_stats_kernel<<<1, 256, 0, stream>>>(stats);
    build_wfrag_kernel<<<1, 256, 0, stream>>>(w_gate, w_scale, w_shift, gW1, gW2);
    zero_h16p_kernel<<<(int)((n8 + 255) / 256), 256, 0, stream>>>((v8h*)h16p, n8);

    conv1_wmma_kernel<<<BB * TT * HH, 320, 0, stream>>>(
        inputs, gW1, g16, stats, stats + 32);

    finalize_stats_kernel<<<1, 32, 0, stream>>>(
        stats, stats + 32, stats + 64, stats + 96, 32, inv_count);

    scan_kernel<<<(BB * 2 * HW) / 256, 256, 0, stream>>>(
        g16, stats + 64, stats + 96, gamma_gate, beta_gate, reverse, h16p);

    conv2_wmma_kernel<<<BB * TT * HH, 320, 0, stream>>>(
        h16p, gW2, raw, stats + 128);

    finalize_stats_kernel<<<1, 32, 0, stream>>>(
        stats + 128, stats + 130, stats + 132, stats + 134, 2, inv_count);

    final_kernel<<<(NPOS + 255) / 256, 256, 0, stream>>>(
        raw, stats + 128, gamma_scale, beta_scale,
        gamma_shift, beta_shift, out);
}
